// BigramSegLayer_62002147885410
// MI455X (gfx1250) — matrix operand order repaired and checked by reading, not compiled, not run on hardware
//
#include <hip/hip_runtime.h>
#include <hip/hip_bf16.h>
#include <math.h>

typedef __attribute__((ext_vector_type(16))) __bf16 v16bf;
typedef __attribute__((ext_vector_type(8)))  __bf16 v8bf;
typedef __attribute__((ext_vector_type(8)))  float  v8f;

#define B_   128
#define T_   512
#define D_   1024
#define L_   5
#define H_   2048
#define BT_  (B_ * T_)
#define HC_  (H_ / 16)     // 128 column chunks of 16
#define KC_  (D_ / 32)     // 32 K chunks of 32
#define XS_  1032          // padded LDS X-tile row stride (bf16); 516 words, %64==4
#define ZS_  19            // padded Z-tile row stride (floats), conflict-free
#define TBL_STRIDE_ 32     // logits6 table row stride (30 used, float4-aligned)

// ---- async B staging geometry ----
#define SUBKC_        8                              // K-steps per async chunk
#define BFRAG_BYTES_  48                             // 32B fragment + 16B pad (16B aligned)
#define BCHUNK_BYTES_ (SUBKC_ * 32 * BFRAG_BYTES_)   // 12288 B per chunk
#define NCHUNKS_      (HC_ * KC_ / SUBKC_)           // 512 chunks (whole W1 stream)

__device__ __forceinline__ float fast_tanh(float x) {
#if __has_builtin(__builtin_amdgcn_tanhf)
  return __builtin_amdgcn_tanhf(x);   // gfx1250 v_tanh_f32 (TRANS op)
#else
  return tanhf(x);
#endif
}

__device__ __forceinline__ void wait_async16() {
  asm volatile("s_wait_asynccnt 0x10" ::: "memory");
}
__device__ __forceinline__ void wait_async0() {
  asm volatile("s_wait_asynccnt 0x0" ::: "memory");
}

// Issue one 12KB B chunk: 16 x global_load_async_to_lds_b128 per wave.
// Global stream is linear; LDS destination is fragment-padded (48B per lane-frag).
__device__ __forceinline__ void async_b_chunk(const __bf16* __restrict__ gbase,
                                              unsigned lds_base, int lane) {
  #pragma unroll
  for (int i = 0; i < 16; ++i) {
    int u = i * 32 + lane;                       // 16B unit index within chunk
    unsigned lds = lds_base +
        (unsigned)((u >> 6) * (32 * BFRAG_BYTES_) +   // kcLocal
                   ((u >> 1) & 31) * BFRAG_BYTES_ +   // fragment lane
                   (u & 1) * 16);                     // low/high 16B
    const void* g = (const char*)gbase + (size_t)u * 16;
    asm volatile("global_load_async_to_lds_b128 %0, %1, off"
                 :: "v"(lds), "v"(g) : "memory");
  }
}

// ---------------------------------------------------------------------------
// Pack W1[0:1024, :] (f32 [IN,H] row-major) into bf16 in WMMA B-fragment order:
// index = ((hc*KC + kc)*32 + lane)*16 + e, lane: N = lane&15, K = kc*32+(lane>>4)*16+e
// ---------------------------------------------------------------------------
__global__ void pack_w1_kernel(const float* __restrict__ W1,
                               __bf16* __restrict__ W1bf) {
  int idx = blockIdx.x * blockDim.x + threadIdx.x;
  if (idx >= HC_ * KC_ * 32 * 16) return;
  int e    = idx & 15;
  int lane = (idx >> 4) & 31;
  int kc   = (idx >> 9) & 31;
  int hc   = idx >> 14;
  int K    = kc * 32 + (lane >> 4) * 16 + e;
  int col  = hc * 16 + (lane & 15);
  W1bf[idx] = (__bf16)W1[(size_t)K * H_ + col];
}

// ---------------------------------------------------------------------------
// Fused: Z = X @ W1[:1024] (bf16 WMMA, f32 acc, dual accumulation chains,
// B double-buffered via async-to-LDS), then for 6 feedback states s:
// logits6[bt, s, l] = sum_h tanh(Z + b1 + u_s) * W2[h,l]  (+ b2).
// One wave per 16 rows of bt = b*T + t.
// ---------------------------------------------------------------------------
__global__ __launch_bounds__(32)
void fused_mlp_wmma_kernel(const float* __restrict__ X,
                           const float* __restrict__ W1,
                           const float* __restrict__ b1,
                           const float* __restrict__ W2,
                           const float* __restrict__ b2,
                           const __bf16* __restrict__ W1bf,
                           float* __restrict__ tbl) {
  __shared__ __align__(16) __bf16 Xt[16 * XS_];                // 33 KB X tile
  __shared__ __align__(16) unsigned char Bb[2 * BCHUNK_BYTES_]; // 24 KB B dbl-buf
  __shared__ float Zt[16 * ZS_];                                // 1.2 KB staging

  const int lane = threadIdx.x;      // 0..31
  const int half = lane >> 4;        // 0 | 1
  const int l16  = lane & 15;
  const int row0 = blockIdx.x * 16;  // base bt row

  const unsigned Bb_base = (unsigned)(size_t)(&Bb[0]);

  // kick off B chunk 0 immediately; it lands while we stage X below
  async_b_chunk(W1bf, Bb_base, lane);

  // ---- stage X tile -> bf16 LDS (coalesced global reads, b64 DS stores) ----
  for (int r = 0; r < 16; ++r) {
    const float* xr = X + (size_t)(row0 + r) * D_;
    #pragma unroll
    for (int it = 0; it < 8; ++it) {
      int c = it * 128 + lane * 4;
      float4 v = *(const float4*)(xr + c);
      __bf16* dst = &Xt[r * XS_ + c];
      dst[0] = (__bf16)v.x; dst[1] = (__bf16)v.y;
      dst[2] = (__bf16)v.z; dst[3] = (__bf16)v.w;
    }
  }
  __syncthreads();

  float acc[6][5] = {};

  for (int hc = 0; hc < HC_; ++hc) {
    v8f ce = {0.f, 0.f, 0.f, 0.f, 0.f, 0.f, 0.f, 0.f};  // even-kc chain
    v8f co = {0.f, 0.f, 0.f, 0.f, 0.f, 0.f, 0.f, 0.f};  // odd-kc chain

    for (int sub = 0; sub < KC_ / SUBKC_; ++sub) {
      const int c = hc * (KC_ / SUBKC_) + sub;
      // prefetch next chunk into the other buffer, then wait for current
      if (c + 1 < NCHUNKS_) {
        async_b_chunk(W1bf + (size_t)(c + 1) * SUBKC_ * 32 * 16,
                      Bb_base + ((unsigned)((c + 1) & 1)) * BCHUNK_BYTES_, lane);
        wait_async16();   // current chunk done; next still in flight
      } else {
        wait_async0();
      }
      const unsigned char* bbuf = &Bb[(c & 1) * BCHUNK_BYTES_];

      #pragma unroll
      for (int kk = 0; kk < SUBKC_; ++kk) {
        const int kc = sub * SUBKC_ + kk;
        // A fragment: row = l16, K = kc*32 + half*16 + {0..15}
        const v8bf* ap = (const v8bf*)&Xt[l16 * XS_ + kc * 32 + half * 16];
        v8bf alo = ap[0];
        v8bf ahi = ap[1];  // +8 elements
        v16bf a = __builtin_shufflevector(alo, ahi,
                    0, 1, 2, 3, 4, 5, 6, 7, 8, 9, 10, 11, 12, 13, 14, 15);
        // B fragment from padded LDS chunk
        const unsigned char* bp = bbuf + (kk * 32 + lane) * BFRAG_BYTES_;
        v8bf blo = *(const v8bf*)(bp);
        v8bf bhi = *(const v8bf*)(bp + 16);
        v16bf bb = __builtin_shufflevector(blo, bhi,
                    0, 1, 2, 3, 4, 5, 6, 7, 8, 9, 10, 11, 12, 13, 14, 15);
        if (kk & 1)
          co = __builtin_amdgcn_wmma_f32_16x16x32_bf16(
                   false, a, false, bb, (short)0, co, false, false);
        else
          ce = __builtin_amdgcn_wmma_f32_16x16x32_bf16(
                   false, a, false, bb, (short)0, ce, false, false);
      }
    }

    // merge the two accumulation chains
    v8f cz;
    #pragma unroll
    for (int r = 0; r < 8; ++r) cz[r] = ce[r] + co[r];

    // ---- transpose C tile through LDS: lane layout -> row ownership ----
    #pragma unroll
    for (int r = 0; r < 8; ++r)
      Zt[(r + 8 * half) * ZS_ + l16] = cz[r];
    __syncthreads();

    // ---- 6-state tanh + W2 reduction; lane owns row l16, n-range by half ----
    #pragma unroll
    for (int i = 0; i < 8; ++i) {
      int n = half * 8 + i;
      int h = hc * 16 + n;
      float b1h = b1[h];                       // same h within each half-wave
      float us[5];
      #pragma unroll
      for (int j = 0; j < 5; ++j)
        us[j] = W1[(size_t)(D_ + j) * H_ + h];
      float w2v[5];
      #pragma unroll
      for (int l = 0; l < 5; ++l)
        w2v[l] = W2[h * 5 + l];

      float zb = Zt[l16 * ZS_ + n] + b1h;
      float tt = fast_tanh(zb);                // state 0: zero feedback
      #pragma unroll
      for (int l = 0; l < 5; ++l) acc[0][l] += tt * w2v[l];
      #pragma unroll
      for (int j = 0; j < 5; ++j) {
        tt = fast_tanh(zb + us[j]);
        #pragma unroll
        for (int l = 0; l < 5; ++l) acc[j + 1][l] += tt * w2v[l];
      }
    }
    __syncthreads();
  }

  // combine the two n-halves (lane r and lane r+16 hold partials for row r)
  #pragma unroll
  for (int s = 0; s < 6; ++s)
    #pragma unroll
    for (int l = 0; l < 5; ++l)
      acc[s][l] += __shfl_xor(acc[s][l], 16, 32);

  if (lane < 16) {
    float* out = tbl + (size_t)(row0 + lane) * TBL_STRIDE_;
    #pragma unroll
    for (int s = 0; s < 6; ++s)
      #pragma unroll
      for (int l = 0; l < 5; ++l)
        out[s * 5 + l] = acc[s][l] + b2[l];
  }
}

// ---------------------------------------------------------------------------
// Sequential greedy scan: 128 independent chains; loads are state-independent
// (all 6 states fetched per step) so only a short ALU chain is serialized.
// ---------------------------------------------------------------------------
__global__ void scan_kernel(const float* __restrict__ tbl,
                            float* __restrict__ logits_out,
                            float* __restrict__ probs_out,
                            int* __restrict__ preds_out,
                            int* __restrict__ oh_out) {
  int b = threadIdx.x;
  if (b >= B_) return;
  int s = 0;  // state 0 = zero one-hot; state j+1 = one-hot(j)
  for (int t = 0; t < T_; ++t) {
    const float4* row = (const float4*)(tbl + (size_t)(b * T_ + t) * TBL_STRIDE_);
    float a[32];
    #pragma unroll
    for (int i = 0; i < 8; ++i) {
      float4 v = row[i];
      a[4 * i + 0] = v.x; a[4 * i + 1] = v.y;
      a[4 * i + 2] = v.z; a[4 * i + 3] = v.w;
    }
    // select the 5 logits of the current state with cndmasks (stay in VGPRs)
    float lg[5];
    #pragma unroll
    for (int l = 0; l < 5; ++l) {
      float v = a[l];
      #pragma unroll
      for (int ss = 1; ss < 6; ++ss)
        v = (s == ss) ? a[ss * 5 + l] : v;
      lg[l] = v;
    }
    // argmax (first-max wins, matching jnp.argmax)
    int p = 0; float m = lg[0];
    #pragma unroll
    for (int l = 1; l < 5; ++l)
      if (lg[l] > m) { m = lg[l]; p = l; }
    // softmax
    float e[5], sum = 0.f;
    #pragma unroll
    for (int l = 0; l < 5; ++l) { e[l] = __expf(lg[l] - m); sum += e[l]; }
    float inv = 1.0f / sum;

    size_t o5 = (size_t)(b * T_ + t) * 5;
    #pragma unroll
    for (int l = 0; l < 5; ++l) {
      logits_out[o5 + l] = lg[l];
      probs_out[o5 + l]  = e[l] * inv;
      oh_out[o5 + l]     = (l == p) ? 1 : 0;
    }
    preds_out[b * T_ + t] = p;
    s = p + 1;
  }
}

// ---------------------------------------------------------------------------
extern "C" void kernel_launch(void* const* d_in, const int* in_sizes, int n_in,
                              void* d_out, int out_size, void* d_ws, size_t ws_size,
                              hipStream_t stream) {
  const float* X  = (const float*)d_in[0];
  // d_in[1] = lengths (unused by the reference outputs)
  const float* W1 = (const float*)d_in[2];
  const float* b1 = (const float*)d_in[3];
  const float* W2 = (const float*)d_in[4];
  const float* b2 = (const float*)d_in[5];

  const size_t w1bf_elems = (size_t)HC_ * KC_ * 32 * 16;              // 2M bf16 = 4 MB
  __bf16* W1bf = (__bf16*)d_ws;
  float*  tbl  = (float*)((char*)d_ws + w1bf_elems * sizeof(__bf16)); // 8 MB table

  pack_w1_kernel<<<(int)((w1bf_elems + 255) / 256), 256, 0, stream>>>(W1, W1bf);
  fused_mlp_wmma_kernel<<<BT_ / 16, 32, 0, stream>>>(X, W1, b1, W2, b2, W1bf, tbl);

  float* logits = (float*)d_out;
  float* probs  = logits + (size_t)BT_ * 5;
  int*   preds  = (int*)(probs + (size_t)BT_ * 5);
  int*   ohs    = preds + BT_;
  scan_kernel<<<1, 128, 0, stream>>>(tbl, logits, probs, preds, ohs);
}
